// DeprelAttention_30039001268276
// MI455X (gfx1250) — matrix operand order repaired and checked by reading, not compiled, not run on hardware
//
#include <hip/hip_runtime.h>
#include <cstdint>

typedef __attribute__((ext_vector_type(16))) _Float16 v16h;
typedef __attribute__((ext_vector_type(8)))  _Float16 v8h;
typedef __attribute__((ext_vector_type(8)))  float    v8f;

#define NEGV (-1e18f)

constexpr int B_  = 16;
constexpr int L_  = 384;
constexpr int D_  = 768;
constexpr int R_  = 768;
constexpr int H_  = 12;
constexpr int DH_ = 64;
constexpr int ROWS = B_ * L_;          // 6144
constexpr int OUT0 = ROWS * D_;        // 4718592 floats (first output)

static __device__ inline v16h make16(v8h lo, v8h hi) {
  return __builtin_shufflevector(lo, hi, 0,1,2,3,4,5,6,7,8,9,10,11,12,13,14,15);
}

static __device__ inline v8f wmma_f16(v16h a, v16h b, v8f c) {
  // D = A(16x32 f16) * B(32x16 f16) + C(16x16 f32)
  return __builtin_amdgcn_wmma_f32_16x16x32_f16(false, a, false, b, (short)0, c, false, false);
}

static __device__ inline v8f zero8() {
  v8f z = {0.f,0.f,0.f,0.f,0.f,0.f,0.f,0.f};
  return z;
}

// ---------------------------------------------------------------------------
// K0: fold Vg into Wgr / Wge:  wgrv[r][h] = sum_j Wgr[r][h*64+j]*Vg[h*64+j]
// ---------------------------------------------------------------------------
__global__ void k_prep(const float* __restrict__ Wgr, const float* __restrict__ Wge,
                       const float* __restrict__ Vg,
                       float* __restrict__ wgrv, float* __restrict__ wgev) {
  int t = blockIdx.x * 256 + threadIdx.x;
  if (t >= 2 * R_ * H_) return;
  int which = t / (R_ * H_);
  int rem   = t - which * (R_ * H_);
  int r = rem / H_;
  int h = rem - r * H_;
  const float* Wm = which ? Wge : Wgr;
  float acc = 0.f;
  #pragma unroll 8
  for (int j = 0; j < DH_; ++j) acc += Wm[r * R_ + h * DH_ + j] * Vg[h * DH_ + j];
  (which ? wgev : wgrv)[r * H_ + h] = acc;
}

// ---------------------------------------------------------------------------
// K1: per (b,l): g[b,h,l] = tanh( rel_row . wgrv + hv_row . wgev ),
//     rlog[b,h,l] = rel_row[h*64..] . Vr[h*64..]
// One wave per row; 4 waves per block of 128 threads.
// ---------------------------------------------------------------------------
__global__ __launch_bounds__(128)
void k_gates(const float* __restrict__ relations, const float* __restrict__ values,
             const int* __restrict__ dp, const uint8_t* __restrict__ rel_mask,
             const float* __restrict__ wgrv, const float* __restrict__ wgev,
             const float* __restrict__ Vr,
             float* __restrict__ gbuf, float* __restrict__ rlbuf) {
  int wid = threadIdx.x >> 5, lane = threadIdx.x & 31;
  int row = blockIdx.x * 4 + wid;               // 0..6143
  int b = row / L_, l = row - b * L_;
  bool rm = rel_mask[b * L_ + l] != 0;
  int idx0 = dp[b * 2 * L_ + l];
  bool hvok = (!rm) && (l != 0);
  const float* relrow = relations + (size_t)(b * L_ + l) * R_;
  const float* hvrow  = values    + (size_t)(b * L_ + idx0) * D_;

  float accg[H_];
  float accr[H_];
  #pragma unroll
  for (int h = 0; h < H_; ++h) { accg[h] = 0.f; accr[h] = 0.f; }

  if (!rm) {
    for (int r = lane; r < R_; r += 32) {
      float rv  = relrow[r];
      float hvv = hvok ? hvrow[r] : 0.f;
      const float4* pg = (const float4*)(wgrv + r * H_);
      const float4* pe = (const float4*)(wgev + r * H_);
      float4 g0 = pg[0], g1 = pg[1], g2 = pg[2];
      float4 e0 = pe[0], e1 = pe[1], e2 = pe[2];
      accg[0]  += rv * g0.x + hvv * e0.x;
      accg[1]  += rv * g0.y + hvv * e0.y;
      accg[2]  += rv * g0.z + hvv * e0.z;
      accg[3]  += rv * g0.w + hvv * e0.w;
      accg[4]  += rv * g1.x + hvv * e1.x;
      accg[5]  += rv * g1.y + hvv * e1.y;
      accg[6]  += rv * g1.z + hvv * e1.z;
      accg[7]  += rv * g1.w + hvv * e1.w;
      accg[8]  += rv * g2.x + hvv * e2.x;
      accg[9]  += rv * g2.y + hvv * e2.y;
      accg[10] += rv * g2.z + hvv * e2.z;
      accg[11] += rv * g2.w + hvv * e2.w;
    }
    #pragma unroll
    for (int h = 0; h < H_; ++h) {
      float a = 0.f;
      for (int j = lane; j < DH_; j += 32)
        a += relrow[h * DH_ + j] * Vr[h * DH_ + j];
      accr[h] = a;
    }
  }
  #pragma unroll
  for (int h = 0; h < H_; ++h) {
    float a = accg[h], c = accr[h];
    for (int m = 16; m; m >>= 1) { a += __shfl_xor(a, m, 32); c += __shfl_xor(c, m, 32); }
    if (lane == 0) {
      gbuf [(b * H_ + h) * L_ + l] = tanhf(a);
      rlbuf[(b * H_ + h) * L_ + l] = c;
    }
  }
}

// ---------------------------------------------------------------------------
// K2: QKV projections.  X(6144x768 f32) @ W(768x768 f32) -> f16 (B,H,L,64).
// 256 thr (8 waves), block tile 128(M) x 64(N), K-step 32, double-buffered LDS,
// vectorized (b128) staging.  blockIdx.y == head; z selects Q/K/V; Q scaled.
// ---------------------------------------------------------------------------
__global__ __launch_bounds__(256)
void k_gemm_qkv(const float* __restrict__ Xq, const float* __restrict__ Xk,
                const float* __restrict__ Xv,
                const float* __restrict__ Wq, const float* __restrict__ Wk,
                const float* __restrict__ Wv,
                _Float16* __restrict__ qb, _Float16* __restrict__ kb,
                _Float16* __restrict__ vb) {
  __shared__ _Float16 Xs[2][128 * 40];
  __shared__ _Float16 Ws[2][64 * 40];
  int z = blockIdx.z;
  const float* X = (z == 0) ? Xq : (z == 1) ? Xk : Xv;
  const float* W = (z == 0) ? Wq : (z == 1) ? Wk : Wv;
  _Float16*  Out = (z == 0) ? qb : (z == 1) ? kb : vb;
  float sc = (z == 0) ? 0.125f : 1.0f;        // 64^-0.5 for Q

  int m0 = blockIdx.x * 128;
  int h  = blockIdx.y;
  int n0 = h * 64;
  int tid = threadIdx.x, wid = tid >> 5, lane = tid & 31;
  int gp = lane >> 4, ln15 = lane & 15;
  int wm0 = (wid & 3) * 32, wn0 = (wid >> 2) * 32;

  auto stage = [&](int buf, int kk) {
    { // X tile 128x32 -> f16, two b128 stores per thread
      int row = tid >> 1, cb = (tid & 1) * 16;
      const float4* p = (const float4*)(X + (size_t)(m0 + row) * 768 + kk + cb);
      float4 a0 = p[0], a1 = p[1], a2 = p[2], a3 = p[3];
      v8f x0 = {a0.x,a0.y,a0.z,a0.w, a1.x,a1.y,a1.z,a1.w};
      v8f x1 = {a2.x,a2.y,a2.z,a2.w, a3.x,a3.y,a3.z,a3.w};
      *(v8h*)(&Xs[buf][row * 40 + cb])     = __builtin_convertvector(x0, v8h);
      *(v8h*)(&Xs[buf][row * 40 + cb + 8]) = __builtin_convertvector(x1, v8h);
    }
    { // W tile transposed Ws[n][k]; thread owns 8 consecutive k of one n
      int n = tid & 63, kb8 = (tid >> 6) * 8;
      const float* wp = W + (size_t)(kk + kb8) * 768 + n0 + n;
      v8f wv;
      #pragma unroll
      for (int j = 0; j < 8; ++j) wv[j] = wp[(size_t)j * 768];
      *(v8h*)(&Ws[buf][n * 40 + kb8]) = __builtin_convertvector(wv, v8h);
    }
  };

  v8f acc[2][2];
  acc[0][0] = zero8(); acc[0][1] = zero8(); acc[1][0] = zero8(); acc[1][1] = zero8();

  stage(0, 0);
  __syncthreads();
  for (int it = 0; it < 24; ++it) {
    int cur = it & 1;
    if (it + 1 < 24) stage(cur ^ 1, (it + 1) * 32);

    const _Float16* xb = &Xs[cur][0];
    const _Float16* wb = &Ws[cur][0];
    v16h a[2], bf[2];
    #pragma unroll
    for (int rt = 0; rt < 2; ++rt) {
      int m = wm0 + rt * 16 + ln15;
      v8h lo = *(const v8h*)(xb + m * 40 + 8 * gp);
      v8h hi = *(const v8h*)(xb + m * 40 + 16 + 8 * gp);
      a[rt] = make16(lo, hi);
    }
    #pragma unroll
    for (int ct = 0; ct < 2; ++ct) {
      int n = wn0 + ct * 16 + ln15;
      v8h lo = *(const v8h*)(wb + n * 40 + 8 * gp);
      v8h hi = *(const v8h*)(wb + n * 40 + 16 + 8 * gp);
      bf[ct] = make16(lo, hi);
    }
    #pragma unroll
    for (int rt = 0; rt < 2; ++rt)
      #pragma unroll
      for (int ct = 0; ct < 2; ++ct)
        acc[rt][ct] = wmma_f16(a[rt], bf[ct], acc[rt][ct]);
    __syncthreads();
  }

  #pragma unroll
  for (int rt = 0; rt < 2; ++rt)
    #pragma unroll
    for (int ct = 0; ct < 2; ++ct)
      #pragma unroll
      for (int r = 0; r < 8; ++r) {
        int ml = wm0 + rt * 16 + r + 8 * gp;
        int nl = wn0 + ct * 16 + ln15;
        int row = m0 + ml;
        int b = row / L_, l = row - b * L_;
        Out[((size_t)(b * H_ + h) * L_ + l) * DH_ + nl] = (_Float16)(acc[rt][ct][r] * sc);
      }
}

// ---------------------------------------------------------------------------
// K4: out = ctx(6144x768 f16) @ Wo(768x768 f32) -> f32 (flat), same scheme.
// ---------------------------------------------------------------------------
__global__ __launch_bounds__(256)
void k_gemm_wo(const _Float16* __restrict__ X, const float* __restrict__ W,
               float* __restrict__ Out) {
  __shared__ _Float16 Xs[2][128 * 40];
  __shared__ _Float16 Ws[2][64 * 40];
  int m0 = blockIdx.x * 128;
  int n0 = blockIdx.y * 64;
  int tid = threadIdx.x, wid = tid >> 5, lane = tid & 31;
  int gp = lane >> 4, ln15 = lane & 15;
  int wm0 = (wid & 3) * 32, wn0 = (wid >> 2) * 32;

  auto stage = [&](int buf, int kk) {
    {
      int row = tid >> 1, cb = (tid & 1) * 16;
      const v8h* p = (const v8h*)(X + (size_t)(m0 + row) * 768 + kk + cb);
      v8h x0 = p[0], x1 = p[1];
      *(v8h*)(&Xs[buf][row * 40 + cb])     = x0;
      *(v8h*)(&Xs[buf][row * 40 + cb + 8]) = x1;
    }
    {
      int n = tid & 63, kb8 = (tid >> 6) * 8;
      const float* wp = W + (size_t)(kk + kb8) * 768 + n0 + n;
      v8f wv;
      #pragma unroll
      for (int j = 0; j < 8; ++j) wv[j] = wp[(size_t)j * 768];
      *(v8h*)(&Ws[buf][n * 40 + kb8]) = __builtin_convertvector(wv, v8h);
    }
  };

  v8f acc[2][2];
  acc[0][0] = zero8(); acc[0][1] = zero8(); acc[1][0] = zero8(); acc[1][1] = zero8();

  stage(0, 0);
  __syncthreads();
  for (int it = 0; it < 24; ++it) {
    int cur = it & 1;
    if (it + 1 < 24) stage(cur ^ 1, (it + 1) * 32);

    const _Float16* xb = &Xs[cur][0];
    const _Float16* wb = &Ws[cur][0];
    v16h a[2], bf[2];
    #pragma unroll
    for (int rt = 0; rt < 2; ++rt) {
      int m = wm0 + rt * 16 + ln15;
      v8h lo = *(const v8h*)(xb + m * 40 + 8 * gp);
      v8h hi = *(const v8h*)(xb + m * 40 + 16 + 8 * gp);
      a[rt] = make16(lo, hi);
    }
    #pragma unroll
    for (int ct = 0; ct < 2; ++ct) {
      int n = wn0 + ct * 16 + ln15;
      v8h lo = *(const v8h*)(wb + n * 40 + 8 * gp);
      v8h hi = *(const v8h*)(wb + n * 40 + 16 + 8 * gp);
      bf[ct] = make16(lo, hi);
    }
    #pragma unroll
    for (int rt = 0; rt < 2; ++rt)
      #pragma unroll
      for (int ct = 0; ct < 2; ++ct)
        acc[rt][ct] = wmma_f16(a[rt], bf[ct], acc[rt][ct]);
    __syncthreads();
  }

  #pragma unroll
  for (int rt = 0; rt < 2; ++rt)
    #pragma unroll
    for (int ct = 0; ct < 2; ++ct)
      #pragma unroll
      for (int r = 0; r < 8; ++r) {
        int ml = wm0 + rt * 16 + r + 8 * gp;
        int nl = wn0 + ct * 16 + ln15;
        Out[(size_t)(m0 + ml) * 768 + n0 + nl] = acc[rt][ct][r];
      }
}

// ---------------------------------------------------------------------------
// K3: fused attention. One block per (b, 32-query tile); loops all 12 heads.
// Dynamic LDS (~226 KB): S / RL(->Vt) / G / ATTW (each 32x384 f32) + WB + QT.
// All (L,L)-shaped tensors live only in LDS.
// ---------------------------------------------------------------------------
static __device__ inline void row_softmax384(float* __restrict__ row, int lane) {
  float x[12];
  float mx = -3.0e38f;
  #pragma unroll
  for (int i = 0; i < 12; ++i) { x[i] = row[lane + 32 * i]; mx = fmaxf(mx, x[i]); }
  for (int m = 16; m; m >>= 1) mx = fmaxf(mx, __shfl_xor(mx, m, 32));
  float s = 0.f;
  #pragma unroll
  for (int i = 0; i < 12; ++i) { x[i] = __expf(x[i] - mx); s += x[i]; }
  for (int m = 16; m; m >>= 1) s += __shfl_xor(s, m, 32);
  float inv = 1.f / s;
  #pragma unroll
  for (int i = 0; i < 12; ++i) row[lane + 32 * i] = x[i] * inv;
}

__global__ __launch_bounds__(256)
void k_attn(const _Float16* __restrict__ qb, const _Float16* __restrict__ kb,
            const _Float16* __restrict__ vb, const uint8_t* __restrict__ mask,
            const int* __restrict__ dp, const float* __restrict__ gbuf,
            const float* __restrict__ rlbuf,
            _Float16* __restrict__ ctx, float* __restrict__ attw_out) {
  extern __shared__ char smem[];
  float*    Sl = (float*)(smem);                 // 32x384 f32  (logits -> sw)
  float*    RL = (float*)(smem + 49152);         // 32x384 f32  (rel logits -> rw)
  _Float16* Vt = (_Float16*)(smem + 49152);      // alias: 64x384 f16 (V^T), after RL dead
  float*    G  = (float*)(smem + 98304);         // 32x384 f32  (gated)
  float*    AW = (float*)(smem + 147456);        // 32x384 f32  (attn-weight head sum)
  _Float16* WB = (_Float16*)(smem + 196608);     // 32x384 f16  (combined weights)
  _Float16* QT = (_Float16*)(smem + 221184);     // 32x72  f16  (Q tile, padded)

  int qt = blockIdx.x, b = blockIdx.y, q0 = qt * 32;
  int tid = threadIdx.x, wid = tid >> 5, lane = tid & 31;
  int gp = lane >> 4, ln15 = lane & 15;
  const v8f vzero = zero8();
  v8f vones; 
  #pragma unroll
  for (int j = 0; j < 8; ++j) vones[j] = 1.f;

  for (int h = 0; h < H_; ++h) {
    const _Float16* Qh = qb + (size_t)(b * H_ + h) * L_ * DH_;
    const _Float16* Kh = kb + (size_t)(b * H_ + h) * L_ * DH_;
    const _Float16* Vh = vb + (size_t)(b * H_ + h) * L_ * DH_;

    // phase A: load Q tile, zero G/RL (vector b128/b256 traffic)
    {
      int idx = tid * 8;
      int row = idx >> 6, c = idx & 63;
      *(v8h*)(QT + row * 72 + c) = *(const v8h*)(Qh + (size_t)(q0 + row) * DH_ + c);
    }
    #pragma unroll
    for (int it = 0; it < 6; ++it) {
      int base = (tid + it * 256) * 8;
      *(v8f*)(G + base)  = vzero;
      *(v8f*)(RL + base) = vzero;
    }
    __syncthreads();

    // phase B: scatter g / rlog via dp_map (s + s^T), and S = Q*K^T (WMMA)
    for (int l = tid; l < L_; l += 256) {
      int p0 = dp[b * 2 * L_ + l];
      int p1 = dp[b * 2 * L_ + L_ + l];
      float gv = gbuf [(b * H_ + h) * L_ + l];
      float rv = rlbuf[(b * H_ + h) * L_ + l];
      if ((unsigned)(p0 - q0) < 32u) { G[(p0 - q0) * L_ + p1] = gv;  RL[(p0 - q0) * L_ + p1] = rv;  }
      if ((unsigned)(p1 - q0) < 32u) { G[(p1 - q0) * L_ + p0] += gv; RL[(p1 - q0) * L_ + p0] += rv; }
    }
    {
      v8f sacc[2][3];
      #pragma unroll
      for (int rt = 0; rt < 2; ++rt)
        #pragma unroll
        for (int c = 0; c < 3; ++c) sacc[rt][c] = zero8();
      #pragma unroll
      for (int k0 = 0; k0 < 64; k0 += 32) {
        v16h a[2];
        #pragma unroll
        for (int rt = 0; rt < 2; ++rt) {
          int m = rt * 16 + ln15;
          v8h lo = *(const v8h*)(QT + m * 72 + k0 + 8 * gp);
          v8h hi = *(const v8h*)(QT + m * 72 + k0 + 16 + 8 * gp);
          a[rt] = make16(lo, hi);
        }
        #pragma unroll
        for (int c = 0; c < 3; ++c) {
          int n = (wid * 3 + c) * 16 + ln15;
          v8h lo = *(const v8h*)(Kh + (size_t)n * DH_ + k0 + 8 * gp);
          v8h hi = *(const v8h*)(Kh + (size_t)n * DH_ + k0 + 16 + 8 * gp);
          v16h bf = make16(lo, hi);
          #pragma unroll
          for (int rt = 0; rt < 2; ++rt)
            sacc[rt][c] = wmma_f16(a[rt], bf, sacc[rt][c]);
        }
      }
      #pragma unroll
      for (int rt = 0; rt < 2; ++rt)
        #pragma unroll
        for (int c = 0; c < 3; ++c)
          #pragma unroll
          for (int r = 0; r < 8; ++r)
            Sl[(rt * 16 + r + 8 * gp) * L_ + (wid * 3 + c) * 16 + ln15] = sacc[rt][c][r];
    }
    __syncthreads();

    // mask + attention_weights head-sum + mask rel logits (8-wide vectors)
    #pragma unroll
    for (int it = 0; it < 6; ++it) {
      int base = (tid + it * 256) * 8;
      int q = base / L_, k = base - q * L_;
      uint64_t m8 = *(const uint64_t*)(mask + ((size_t)b * L_ + q0 + q) * L_ + k);
      v8f s  = *(v8f*)(Sl + base);
      v8f rl = *(v8f*)(RL + base);
      v8f aw = (h == 0) ? vzero : *(v8f*)(AW + base);
      #pragma unroll
      for (int j = 0; j < 8; ++j) {
        bool mv = ((m8 >> (8 * j)) & 0xffull) != 0;
        float sv = s[j];
        if (mv) { sv = NEGV; rl[j] = NEGV; }
        s[j] = sv;
        aw[j] += sv * (1.f / 12.f);
      }
      *(v8f*)(Sl + base) = s;
      *(v8f*)(RL + base) = rl;
      *(v8f*)(AW + base) = aw;
    }
    __syncthreads();

    // softmax rows (one wave owns 4 rows)
    #pragma unroll
    for (int rr = 0; rr < 4; ++rr) {
      row_softmax384(Sl + (wid * 4 + rr) * L_, lane);
      row_softmax384(RL + (wid * 4 + rr) * L_, lane);
    }
    __syncthreads();

    // w = (1-g)*sw + g*rw  -> f16 (vectorized)
    #pragma unroll
    for (int it = 0; it < 6; ++it) {
      int base = (tid + it * 256) * 8;
      v8f gv = *(v8f*)(G + base);
      v8f sw = *(v8f*)(Sl + base);
      v8f rw = *(v8f*)(RL + base);
      v8f w  = (vones - gv) * sw + gv * rw;
      *(v8h*)(WB + base) = __builtin_convertvector(w, v8h);
    }
    __syncthreads();

    // stage V^T into (dead) RL region; thread packs 8 k's of one d -> b128
    #pragma unroll
    for (int it = 0; it < 12; ++it) {
      int idx = tid + it * 256;          // 0..3071
      int d = idx & 63, kb8 = idx >> 6;  // kb8: 0..47
      const _Float16* vp = Vh + (size_t)(kb8 * 8) * DH_ + d;
      v8h vv;
      #pragma unroll
      for (int j = 0; j < 8; ++j) vv[j] = vp[(size_t)j * DH_];
      *(v8h*)(Vt + d * L_ + kb8 * 8) = vv;
    }
    __syncthreads();

    // ctx = w @ V  (32x384 @ 384x64) ; 8 waves = 8 output tiles
    {
      int rt = wid & 1, nt = wid >> 1;
      v8f cacc = zero8();
      #pragma unroll 4
      for (int k0 = 0; k0 < L_; k0 += 32) {
        int m = rt * 16 + ln15;
        v8h lo = *(const v8h*)(WB + m * L_ + k0 + 8 * gp);
        v8h hi = *(const v8h*)(WB + m * L_ + k0 + 16 + 8 * gp);
        v16h a = make16(lo, hi);
        int n = nt * 16 + ln15;
        v8h bl = *(const v8h*)(Vt + n * L_ + k0 + 8 * gp);
        v8h bh = *(const v8h*)(Vt + n * L_ + k0 + 16 + 8 * gp);
        v16h bf = make16(bl, bh);
        cacc = wmma_f16(a, bf, cacc);
      }
      #pragma unroll
      for (int r = 0; r < 8; ++r) {
        int m = rt * 16 + r + 8 * gp;
        int n = nt * 16 + ln15;
        ctx[((size_t)(b * L_) + q0 + m) * D_ + h * DH_ + n] = (_Float16)cacc[r];
      }
    }
    __syncthreads();
  }

  // write attention_weights tile (32-byte global stores)
  #pragma unroll
  for (int it = 0; it < 6; ++it) {
    int base = (tid + it * 256) * 8;
    int q = base / L_, k = base - q * L_;
    v8f a = *(v8f*)(AW + base);
    *(v8f*)(attw_out + ((size_t)(b * L_) + q0 + q) * L_ + k) = a;
  }
}

// ---------------------------------------------------------------------------
extern "C" void kernel_launch(void* const* d_in, const int* in_sizes, int n_in,
                              void* d_out, int out_size, void* d_ws, size_t ws_size,
                              hipStream_t stream) {
  (void)in_sizes; (void)n_in; (void)out_size; (void)ws_size;
  const float*   queries   = (const float*)d_in[0];
  const float*   keys      = (const float*)d_in[1];
  const float*   values    = (const float*)d_in[2];
  const float*   relations = (const float*)d_in[3];
  const int*     dp        = (const int*)d_in[4];
  const uint8_t* mask      = (const uint8_t*)d_in[5];
  const uint8_t* rel_mask  = (const uint8_t*)d_in[6];
  const float*   Wq        = (const float*)d_in[7];
  const float*   Wk        = (const float*)d_in[8];
  const float*   Wv        = (const float*)d_in[9];
  const float*   Wo        = (const float*)d_in[10];
  const float*   Wge       = (const float*)d_in[11];
  const float*   Wgr       = (const float*)d_in[12];
  const float*   Vr        = (const float*)d_in[13];
  const float*   Vg        = (const float*)d_in[14];

  char* ws = (char*)d_ws;
  constexpr size_t HBUF = (size_t)B_ * H_ * L_ * DH_ * 2;   // 9,437,184 B (f16)
  _Float16* qbuf  = (_Float16*)(ws);
  _Float16* kbuf  = (_Float16*)(ws + HBUF);
  _Float16* vbuf  = (_Float16*)(ws + 2 * HBUF);
  _Float16* ctx   = (_Float16*)(ws + 3 * HBUF);             // (B*L, 768) f16
  float*    gbuf  = (float*)(ws + 4 * HBUF);                // (B,H,L) f32
  float*    rlbuf = (float*)(ws + 4 * HBUF + 294912);
  float*    wgrv  = (float*)(ws + 4 * HBUF + 2 * 294912);   // (768,12) f32
  float*    wgev  = (float*)(ws + 4 * HBUF + 2 * 294912 + 36864);

  float* out  = (float*)d_out;
  float* attw = out + OUT0;

  k_prep<<<72, 256, 0, stream>>>(Wgr, Wge, Vg, wgrv, wgev);
  k_gates<<<ROWS / 4, 128, 0, stream>>>(relations, values, dp, rel_mask,
                                        wgrv, wgev, Vr, gbuf, rlbuf);
  k_gemm_qkv<<<dim3(ROWS / 128, H_, 3), 256, 0, stream>>>(
      queries, keys, values, Wq, Wk, Wv, qbuf, kbuf, vbuf);

  const int attn_lds = 225792;  // 4x48KB f32 + 24KB WB + 4.5KB QT
  (void)hipFuncSetAttribute((const void*)k_attn,
                            hipFuncAttributeMaxDynamicSharedMemorySize, attn_lds);
  k_attn<<<dim3(L_ / 32, B_), 256, attn_lds, stream>>>(
      qbuf, kbuf, vbuf, mask, dp, gbuf, rlbuf, ctx, attw);

  k_gemm_wo<<<dim3(ROWS / 128, D_ / 64, 1), 256, 0, stream>>>(ctx, Wo, out);
}
